// MAGNAKGE_34299608826413
// MI455X (gfx1250) — matrix operand order repaired and checked by reading, not compiled, not run on hardware
//
#include <hip/hip_runtime.h>
#include <hip/hip_bf16.h>

// ---------------------------------------------------------------------------
// MAGNA-KGE pipeline for gfx1250 (MI455X): bf16 WMMA GEMMs + atomic-free
// CSR-gather graph attention (L2-resident), deterministic across replays.
// ---------------------------------------------------------------------------

typedef __bf16 bf16;
typedef __attribute__((ext_vector_type(16))) bf16  v16bf;
typedef __attribute__((ext_vector_type(8)))  float v8f;

#define NN 204800      // nodes
#define NE 1638400     // edges
#define HID 128
#define HEADS 8
#define HEAD_D 16
#define ALPHA 0.15f
#define NEG_SLOPE 0.2f
#define BATCH 8192
#define SCAN_BLOCKS 200          // NN / 1024
#define CDIV(a,b) (((a)+(b)-1)/(b))

__device__ __forceinline__ bf16 f2bf(float f) {
    unsigned u = __float_as_uint(f);
    u += 0x7FFFu + ((u >> 16) & 1u);          // round-to-nearest-even
    unsigned short h = (unsigned short)(u >> 16);
    return __builtin_bit_cast(bf16, h);
}

// ---------------- elementwise / conversion kernels ----------------

__global__ void k_f2bf(const float* __restrict__ in, bf16* __restrict__ out, size_t n) {
    size_t i = (size_t)blockIdx.x * blockDim.x + threadIdx.x;
    if (i < n) out[i] = f2bf(in[i]);
}

// W[K,N] f32 (row-major) -> WT[N,K] bf16 (row-major)
__global__ void k_transpose_f2bf(const float* __restrict__ W, bf16* __restrict__ WT,
                                 int K, int N) {
    size_t i = (size_t)blockIdx.x * blockDim.x + threadIdx.x;
    if (i >= (size_t)K * N) return;
    int n = (int)(i / K), k = (int)(i % K);
    WT[i] = f2bf(W[(size_t)k * N + n]);
}

__global__ void k_residual(float* __restrict__ out, const float* __restrict__ hfin,
                           const float* __restrict__ Xin, size_t n) {
    size_t i = (size_t)blockIdx.x * blockDim.x + threadIdx.x;
    if (i < n) out[i] = hfin[i] + Xin[i];
}

// ---------------- CSR-by-destination build (once per call) ----------------

__global__ void k_zero_int(int* __restrict__ p, int n) {
    int i = blockIdx.x * blockDim.x + threadIdx.x;
    if (i < n) p[i] = 0;
}

__global__ void k_count(const int* __restrict__ dst, int* __restrict__ deg) {
    int e = blockIdx.x * blockDim.x + threadIdx.x;
    if (e < NE) atomicAdd(&deg[dst[e]], 1);
}

// per-block exclusive scan (1024 elems/block, 256 threads x 4 elems)
__global__ void k_scan1(const int* __restrict__ deg, int* __restrict__ rowptr,
                        int* __restrict__ bsum) {
    __shared__ int sh[256];
    int base = blockIdx.x * 1024;
    int t = threadIdx.x;
    int v[4]; int s = 0;
#pragma unroll
    for (int j = 0; j < 4; ++j) { v[j] = deg[base + t * 4 + j]; s += v[j]; }
    sh[t] = s; __syncthreads();
    for (int off = 1; off < 256; off <<= 1) {
        int x = (t >= off) ? sh[t - off] : 0;
        __syncthreads();
        sh[t] += x;
        __syncthreads();
    }
    if (t == 255) bsum[blockIdx.x] = sh[255];
    int run = (t == 0) ? 0 : sh[t - 1];
#pragma unroll
    for (int j = 0; j < 4; ++j) { rowptr[base + t * 4 + j] = run; run += v[j]; }
}

// serial exclusive scan of block sums; writes rowptr[NN] = total
__global__ void k_scan_top(int* __restrict__ bsum, int* __restrict__ rowptr) {
    if (threadIdx.x == 0 && blockIdx.x == 0) {
        int acc = 0;
        for (int i = 0; i < SCAN_BLOCKS; ++i) { int v = bsum[i]; bsum[i] = acc; acc += v; }
        rowptr[NN] = acc;
    }
}

__global__ void k_scan_add(int* __restrict__ rowptr, const int* __restrict__ bsum) {
    int i = blockIdx.x * blockDim.x + threadIdx.x;
    if (i < NN) rowptr[i] += bsum[i >> 10];
}

__global__ void k_copy_int(int* __restrict__ out, const int* __restrict__ in, int n) {
    int i = blockIdx.x * blockDim.x + threadIdx.x;
    if (i < n) out[i] = in[i];
}

__global__ void k_fill(const int* __restrict__ dst, int* __restrict__ cur,
                       int* __restrict__ eid) {
    int e = blockIdx.x * blockDim.x + threadIdx.x;
    if (e >= NE) return;
    int pos = atomicAdd(&cur[dst[e]], 1);
    eid[pos] = e;
}

// insertion-sort each node's edge list -> deterministic eid order
__global__ void k_sortseg(const int* __restrict__ rowptr, int* __restrict__ eid) {
    int n = blockIdx.x * blockDim.x + threadIdx.x;
    if (n >= NN) return;
    int beg = rowptr[n], end = rowptr[n + 1];
    for (int i = beg + 1; i < end; ++i) {
        int key = eid[i]; int j = i - 1;
        while (j >= beg && eid[j] > key) { eid[j + 1] = eid[j]; --j; }
        eid[j + 1] = key;
    }
}

// ---------------- attention pipeline kernels ----------------

// C[k,h] = sum_d W_rel[k, h*16+d] * a_r[h*16+d]   (32x8, one block)
__global__ void k_relfold(const float* __restrict__ Wrel, const float* __restrict__ a_r,
                          float* __restrict__ C) {
    int t = threadIdx.x;                  // 256 threads
    int k = t >> 3, h = t & 7;
    float acc = 0.f;
    for (int d = 0; d < HEAD_D; ++d)
        acc += Wrel[(size_t)k * HID + h * HEAD_D + d] * a_r[h * HEAD_D + d];
    C[k * HEADS + h] = acc;
}

// es[n,h] = ft[n,h,:].a_s[h,:] ; ed likewise. one thread per node.
__global__ void k_node_logits(const float* __restrict__ ft, const float* __restrict__ a_s,
                              const float* __restrict__ a_d, float* __restrict__ es,
                              float* __restrict__ ed) {
    int n = blockIdx.x * blockDim.x + threadIdx.x;
    if (n >= NN) return;
    const float* f = ft + (size_t)n * HID;
    for (int h = 0; h < HEADS; ++h) {
        float s = 0.f, d = 0.f;
        for (int t = 0; t < HEAD_D; ++t) {
            float v = f[h * HEAD_D + t];
            s += v * a_s[h * HEAD_D + t];
            d += v * a_d[h * HEAD_D + t];
        }
        es[(size_t)n * HEADS + h] = s;
        ed[(size_t)n * HEADS + h] = d;
    }
}

// er[e,h] = r[e,:] @ C[:,h]  -- one thread per edge
__global__ void k_er(const float* __restrict__ rel, const float* __restrict__ C,
                     float* __restrict__ er) {
    int e = blockIdx.x * blockDim.x + threadIdx.x;
    if (e >= NE) return;
    const float* r = rel + (size_t)e * 32;
    for (int h = 0; h < HEADS; ++h) {
        float acc = 0.f;
        for (int k = 0; k < 32; ++k) acc += r[k] * C[k * HEADS + h];
        er[(size_t)e * HEADS + h] = acc;
    }
}

// logits + leaky relu (no atomics)
__global__ void k_edge_logits(const int* __restrict__ src, const int* __restrict__ dst,
                              const float* __restrict__ es, const float* __restrict__ ed,
                              const float* __restrict__ er, float* __restrict__ logit) {
    size_t t = (size_t)blockIdx.x * blockDim.x + threadIdx.x;
    if (t >= (size_t)NE * HEADS) return;
    int e = (int)(t >> 3), h = (int)(t & 7);
    float v = es[(size_t)src[e] * HEADS + h] + ed[(size_t)dst[e] * HEADS + h] + er[t];
    logit[t] = (v >= 0.f) ? v : NEG_SLOPE * v;
}

// segment softmax over incoming edges via CSR; in-place logit -> att
__global__ void k_softmax_csr(const int* __restrict__ rowptr, const int* __restrict__ eid,
                              float* __restrict__ att) {
    int t = blockIdx.x * blockDim.x + threadIdx.x;
    if (t >= NN * HEADS) return;
    int n = t >> 3, h = t & 7;
    int beg = rowptr[n], end = rowptr[n + 1];
    float m = -3.402823466e38f;
    for (int i = beg; i < end; ++i)
        m = fmaxf(m, att[(size_t)eid[i] * HEADS + h]);
    float z = 0.f;
    for (int i = beg; i < end; ++i)
        z += __expf(att[(size_t)eid[i] * HEADS + h] - m);
    float inv = 1.f / (z + 1e-9f);
    for (int i = beg; i < end; ++i) {
        size_t idx = (size_t)eid[i] * HEADS + h;
        att[idx] = __expf(att[idx] - m) * inv;
    }
}

// one diffusion hop, gather form (atomic-free):
// hout[n,h,:] = (1-a) * sum_e att[e,h] * hin[src[e],h,:] + a * h0[n,h,:]
__global__ void k_hop_csr(const int* __restrict__ rowptr, const int* __restrict__ eid,
                          const int* __restrict__ src, const float* __restrict__ att,
                          const float* __restrict__ hin, const float* __restrict__ h0,
                          float* __restrict__ hout) {
    int t = blockIdx.x * blockDim.x + threadIdx.x;
    if (t >= NN * HEADS) return;
    int n = t >> 3, h = t & 7;
    int beg = rowptr[n], end = rowptr[n + 1];
    float4 a0 = {0,0,0,0}, a1 = {0,0,0,0}, a2 = {0,0,0,0}, a3 = {0,0,0,0};
    for (int i = beg; i < end; ++i) {
        int e = eid[i];
        float w = att[(size_t)e * HEADS + h];
        const float4* hs = (const float4*)(hin + (size_t)src[e] * HID + h * HEAD_D);
        float4 v;
        v = hs[0]; a0.x += w*v.x; a0.y += w*v.y; a0.z += w*v.z; a0.w += w*v.w;
        v = hs[1]; a1.x += w*v.x; a1.y += w*v.y; a1.z += w*v.z; a1.w += w*v.w;
        v = hs[2]; a2.x += w*v.x; a2.y += w*v.y; a2.z += w*v.z; a2.w += w*v.w;
        v = hs[3]; a3.x += w*v.x; a3.y += w*v.y; a3.z += w*v.z; a3.w += w*v.w;
    }
    const float4* p0 = (const float4*)(h0 + (size_t)n * HID + h * HEAD_D);
    float4* po = (float4*)(hout + (size_t)n * HID + h * HEAD_D);
    const float oa = 1.f - ALPHA;
    float4 r, b;
    b = p0[0]; r.x = oa*a0.x + ALPHA*b.x; r.y = oa*a0.y + ALPHA*b.y;
               r.z = oa*a0.z + ALPHA*b.z; r.w = oa*a0.w + ALPHA*b.w; po[0] = r;
    b = p0[1]; r.x = oa*a1.x + ALPHA*b.x; r.y = oa*a1.y + ALPHA*b.y;
               r.z = oa*a1.z + ALPHA*b.z; r.w = oa*a1.w + ALPHA*b.w; po[1] = r;
    b = p0[2]; r.x = oa*a2.x + ALPHA*b.x; r.y = oa*a2.y + ALPHA*b.y;
               r.z = oa*a2.z + ALPHA*b.z; r.w = oa*a2.w + ALPHA*b.w; po[2] = r;
    b = p0[3]; r.x = oa*a3.x + ALPHA*b.x; r.y = oa*a3.y + ALPHA*b.y;
               r.z = oa*a3.z + ALPHA*b.z; r.w = oa*a3.w + ALPHA*b.w; po[3] = r;
}

// ---------------- bf16 WMMA GEMM: C[M,N] = A[M,K] @ BT[N,K]^T (+bias, relu) ---
// 4 waves/block; each wave computes a 64x64 tile (4x4 grid of 16x16x32 WMMAs).
// Requires M%128==0, N%128==0, K%32==0.

union AFrag { v16bf v; uint4 q[2]; };

__global__ __launch_bounds__(128)
void k_gemm_bf16(const bf16* __restrict__ A, const bf16* __restrict__ BT,
                 float* __restrict__ C, const float* __restrict__ bias,
                 int M, int N, int K, int relu) {
    const int lane = threadIdx.x & 31;
    const int wave = threadIdx.x >> 5;
    const int wr = wave >> 1, wc = wave & 1;
    const int mBase = blockIdx.y * 128 + wr * 64;
    const int nBase = blockIdx.x * 128 + wc * 64;
    const int half = lane >> 4;     // K-half select per 16-bit WMMA layout
    const int l16  = lane & 15;

    v8f acc[4][4];
    v8f zero = {0.f, 0.f, 0.f, 0.f, 0.f, 0.f, 0.f, 0.f};
#pragma unroll
    for (int i = 0; i < 4; ++i)
#pragma unroll
        for (int j = 0; j < 4; ++j) acc[i][j] = zero;

    for (int k0 = 0; k0 < K; k0 += 32) {
        AFrag a[4], b[4];
#pragma unroll
        for (int i = 0; i < 4; ++i) {
            // A 16-bit layout: lanes 0-15 hold K=k0+0..7 / k0+16..23 of row M=l16,
            //                  lanes 16-31 hold K=k0+8..15 / k0+24..31.
            const bf16* p = A + (size_t)(mBase + i * 16 + l16) * K + k0 + half * 8;
            a[i].q[0] = *(const uint4*)(p);
            a[i].q[1] = *(const uint4*)(p + 16);
        }
#pragma unroll
        for (int j = 0; j < 4; ++j) {
            // B layout: lanes 0-15 hold K=k0+0..7 / k0+16..23 of col N=l16,
            //           lanes 16-31 hold K=k0+8..15 / k0+24..31.  BT is [N,K] row-major.
            const bf16* p = BT + (size_t)(nBase + j * 16 + l16) * K + k0 + half * 8;
            b[j].q[0] = *(const uint4*)(p);
            b[j].q[1] = *(const uint4*)(p + 16);
        }
#pragma unroll
        for (int i = 0; i < 4; ++i)
#pragma unroll
            for (int j = 0; j < 4; ++j)
                acc[i][j] = __builtin_amdgcn_wmma_f32_16x16x32_bf16(
                    false, a[i].v, false, b[j].v, (short)0, acc[i][j], false, false);
    }

    // C/D layout: lane l16 VGPR r = (M = r + half*8, N = l16)
#pragma unroll
    for (int i = 0; i < 4; ++i) {
#pragma unroll
        for (int j = 0; j < 4; ++j) {
            int col = nBase + j * 16 + l16;
            float bv = bias ? bias[col] : 0.f;
#pragma unroll
            for (int r = 0; r < 8; ++r) {
                int row = mBase + i * 16 + half * 8 + r;
                float v = acc[i][j][r] + bv;
                if (relu) v = fmaxf(v, 0.f);
                C[(size_t)row * N + col] = v;
            }
        }
    }
}

// final tiny GEMM: logits[b,j] = h2[b,:] @ W3[:,j] + b3[j]   (N=8)
__global__ void k_final(const float* __restrict__ h2, const float* __restrict__ W3,
                        const float* __restrict__ b3, float* __restrict__ out) {
    int t = blockIdx.x * blockDim.x + threadIdx.x;
    if (t >= BATCH * 8) return;
    int b = t >> 3, j = t & 7;
    const float* hr = h2 + (size_t)b * 640;
    float acc = b3[j];
    for (int k = 0; k < 640; ++k) acc += hr[k] * W3[k * 8 + j];
    out[t] = acc;
}

// ---------------- host orchestration ----------------

static void run_magna_layer(const float* Xin, const float* rel, const int* src,
                            const int* dst, const float* Went, const float* Wrel,
                            const float* a_s, const float* a_d, const float* a_r,
                            float* Xout,
                            bf16* Xbf, float* h0, float* hA, float* hB,
                            float* es, float* ed, float* er, float* att,
                            float* Cfold, bf16* WT,
                            const int* rowptr, const int* eid,
                            hipStream_t s) {
    const size_t nxf = (size_t)NN * HID;
    const size_t neh = (size_t)NE * HEADS;
    const int    nnh = NN * HEADS;

    k_f2bf<<<CDIV(nxf, 256), 256, 0, s>>>(Xin, Xbf, nxf);
    k_transpose_f2bf<<<CDIV(HID * HID, 256), 256, 0, s>>>(Went, WT, HID, HID);
    {   // ft = X @ W_ent  : M=204800 N=128 K=128
        dim3 g(HID / 128, NN / 128);
        k_gemm_bf16<<<g, 128, 0, s>>>(Xbf, WT, h0, nullptr, NN, HID, HID, 0);
    }
    k_relfold<<<1, 256, 0, s>>>(Wrel, a_r, Cfold);
    k_node_logits<<<CDIV(NN, 256), 256, 0, s>>>(h0, a_s, a_d, es, ed);
    k_er<<<CDIV(NE, 256), 256, 0, s>>>(rel, Cfold, er);
    k_edge_logits<<<CDIV(neh, 256), 256, 0, s>>>(src, dst, es, ed, er, att);
    k_softmax_csr<<<CDIV(nnh, 256), 256, 0, s>>>(rowptr, eid, att);

    // 6 hops, gather form, ping-pong (h0 -> hA -> hB -> ... -> hB)
    const float* in = h0;
    float* out = hA;
    for (int hop = 0; hop < 6; ++hop) {
        k_hop_csr<<<CDIV(nnh, 256), 256, 0, s>>>(rowptr, eid, src, att, in, h0, out);
        in  = out;
        out = (out == hA) ? hB : hA;
    }
    k_residual<<<CDIV(nxf, 256), 256, 0, s>>>(Xout, hB, Xin, nxf);
}

extern "C" void kernel_launch(void* const* d_in, const int* in_sizes, int n_in,
                              void* d_out, int out_size, void* d_ws, size_t ws_size,
                              hipStream_t stream) {
    const float* X    = (const float*)d_in[0];
    const float* rel  = (const float*)d_in[1];
    const int*   src  = (const int*)d_in[2];
    const int*   dst  = (const int*)d_in[3];
    const float* We0  = (const float*)d_in[4];
    const float* Wr0  = (const float*)d_in[5];
    const float* as0  = (const float*)d_in[6];
    const float* ad0  = (const float*)d_in[7];
    const float* ar0  = (const float*)d_in[8];
    const float* We1  = (const float*)d_in[9];
    const float* Wr1  = (const float*)d_in[10];
    const float* as1  = (const float*)d_in[11];
    const float* ad1  = (const float*)d_in[12];
    const float* ar1  = (const float*)d_in[13];
    const float* W1   = (const float*)d_in[14];
    const float* b1   = (const float*)d_in[15];
    const float* W2   = (const float*)d_in[16];
    const float* b2   = (const float*)d_in[17];
    const float* W3   = (const float*)d_in[18];
    const float* b3   = (const float*)d_in[19];
    float* out = (float*)d_out;

    // ---- workspace carve-up (256B aligned) ----
    char* p = (char*)d_ws;
    auto carve = [&](size_t bytes) -> char* {
        char* r = p;
        p += (bytes + 255) & ~(size_t)255;
        return r;
    };
    const size_t nxf = (size_t)NN * HID;
    bf16*  Xbf   = (bf16*) carve(nxf * 2);                 // 52.4 MB; MLP input bf
    float* h0    = (float*)carve(nxf * 4);                 // 104.9 MB; reused as h1bf
    float* hA    = (float*)carve(nxf * 4);                 // 104.9 MB; reused as h1
    float* hB    = (float*)carve(nxf * 4);                 // 104.9 MB; reused as h2
    float* xmid  = (float*)carve(nxf * 4);                 // 104.9 MB layer out / MLP in
    float* es    = (float*)carve((size_t)NN * HEADS * 4);
    float* ed    = (float*)carve((size_t)NN * HEADS * 4);
    float* er    = (float*)carve((size_t)NE * HEADS * 4);  // 52.4 MB
    float* att   = (float*)carve((size_t)NE * HEADS * 4);  // 52.4 MB
    float* Cfold = (float*)carve(32 * HEADS * 4);
    bf16*  WT    = (bf16*) carve((size_t)2560 * 3200 * 2); // 16.4 MB (largest W^T)
    int*   deg    = (int*)carve((size_t)NN * 4);
    int*   rowptr = (int*)carve((size_t)(NN + 1) * 4);
    int*   cur    = (int*)carve((size_t)NN * 4);
    int*   eid    = (int*)carve((size_t)NE * 4);           // 6.6 MB
    int*   bsum   = (int*)carve((size_t)SCAN_BLOCKS * 4);

    // ---- build CSR by destination (deterministic after per-segment sort) ----
    k_zero_int<<<CDIV(NN, 256), 256, 0, stream>>>(deg, NN);
    k_count<<<CDIV(NE, 256), 256, 0, stream>>>(dst, deg);
    k_scan1<<<SCAN_BLOCKS, 256, 0, stream>>>(deg, rowptr, bsum);
    k_scan_top<<<1, 64, 0, stream>>>(bsum, rowptr);
    k_scan_add<<<CDIV(NN, 256), 256, 0, stream>>>(rowptr, bsum);
    k_copy_int<<<CDIV(NN, 256), 256, 0, stream>>>(cur, rowptr, NN);
    k_fill<<<CDIV(NE, 256), 256, 0, stream>>>(dst, cur, eid);
    k_sortseg<<<CDIV(NN, 256), 256, 0, stream>>>(rowptr, eid);

    // ---- layer 0: X(d_in) -> xmid ----
    run_magna_layer(X, rel, src, dst, We0, Wr0, as0, ad0, ar0, xmid,
                    Xbf, h0, hA, hB, es, ed, er, att, Cfold, WT,
                    rowptr, eid, stream);
    // ---- layer 1: xmid -> xmid (elementwise residual target, safe in place) ----
    run_magna_layer(xmid, rel, src, dst, We1, Wr1, as1, ad1, ar1, xmid,
                    Xbf, h0, hA, hB, es, ed, er, att, Cfold, WT,
                    rowptr, eid, stream);

    // ---- policy MLP ----
    // H = xmid viewed as [8192, 3200]
    k_f2bf<<<CDIV(nxf, 256), 256, 0, stream>>>(xmid, Xbf, nxf);
    k_transpose_f2bf<<<CDIV(3200 * 2560, 256), 256, 0, stream>>>(W1, WT, 3200, 2560);
    {   // h1 = relu(H @ W1 + b1) : M=8192 N=2560 K=3200  -> hA
        dim3 g(2560 / 128, BATCH / 128);
        k_gemm_bf16<<<g, 128, 0, stream>>>(Xbf, WT, hA, b1, BATCH, 2560, 3200, 1);
    }
    {   // h1 bf16 -> h0 buffer
        size_t n1 = (size_t)BATCH * 2560;
        k_f2bf<<<CDIV(n1, 256), 256, 0, stream>>>(hA, (bf16*)h0, n1);
    }
    k_transpose_f2bf<<<CDIV(2560 * 640, 256), 256, 0, stream>>>(W2, WT, 2560, 640);
    {   // h2 = relu(h1 @ W2 + b2) : M=8192 N=640 K=2560 -> hB
        dim3 g(640 / 128, BATCH / 128);
        k_gemm_bf16<<<g, 128, 0, stream>>>((bf16*)h0, WT, hB, b2, BATCH, 640, 2560, 1);
    }
    // logits = h2 @ W3 + b3
    k_final<<<CDIV(BATCH * 8, 256), 256, 0, stream>>>(hB, W3, b3, out);
    (void)in_sizes; (void)n_in; (void)out_size; (void)ws_size;
}
